// RNNCell_44555990729391
// MI455X (gfx1250) — matrix operand order repaired and checked by reading, not compiled, not run on hardware
//
#include <hip/hip_runtime.h>

// ---------------------------------------------------------------------------
// RNN forward for MI455X (gfx1250, wave32, WMMA).
//   xp = x @ Wih^T + b_ih                  (bf16 WMMA GEMM, f32 out)
//   for t: h = relu(xp_t + h @ Wh^T + b_h) (persistent WMMA kernel, grid sync)
//   y  = hs @ V^T + b_v                    (bf16 WMMA GEMM, f32 out)
// Register double-buffered software pipeline: loads for K-chunk k+1 issue
// before the WMMAs of chunk k, so L2 latency overlaps matrix math.
// ---------------------------------------------------------------------------

typedef __attribute__((ext_vector_type(16))) __bf16 v16bf;
typedef __attribute__((ext_vector_type(8)))  float  v8f;

constexpr int B = 64, S = 512, I = 512, H = 1024;

// Workspace layout (bytes)
constexpr size_t XBF_OFF  = 0;                                   // x as bf16        [B*S, I]
constexpr size_t WIH_OFF  = XBF_OFF  + (size_t)B * S * I * 2;    // Wih bf16         [H, I]
constexpr size_t WH_OFF   = WIH_OFF  + (size_t)H * I * 2;        // Wh  bf16         [H, H]
constexpr size_t VW_OFF   = WH_OFF   + (size_t)H * H * 2;        // V   bf16         [H, H]
constexpr size_t XP_OFF   = VW_OFF   + (size_t)H * H * 2;        // xp  f32          [B, S, H]
constexpr size_t HS_OFF   = XP_OFF   + (size_t)B * S * H * 4;    // hs  bf16         [S, B, H]
constexpr size_t HBUF_OFF = HS_OFF   + (size_t)S * B * H * 2;    // h double buffer  [2, B, H] bf16
constexpr size_t SYNC_OFF = HBUF_OFF + (size_t)2 * B * H * 2;    // grid-sync counter

union ABFrag { uint4 u[2]; v16bf v; };

// Load NP fragments (2 x 16B each) for one 32-wide K chunk starting at kb.
// A and B fragments: per-lane halves are 16 elements apart.
template<int NP>
__device__ __forceinline__ void load_frags(ABFrag (&buf)[NP],
                                           const __bf16* const (&p)[NP],
                                           int kb, int second_off)
{
#pragma unroll
    for (int i = 0; i < NP; ++i) {
        buf[i].u[0] = *reinterpret_cast<const uint4*>(p[i] + kb);
        buf[i].u[1] = *reinterpret_cast<const uint4*>(p[i] + kb + second_off);
    }
}

template<int MF, int NF>
__device__ __forceinline__ void do_wmma(v8f (&acc)[MF][NF],
                                        ABFrag (&a)[MF], ABFrag (&b)[NF])
{
#pragma unroll
    for (int m = 0; m < MF; ++m)
#pragma unroll
        for (int f = 0; f < NF; ++f)
            acc[m][f] = __builtin_amdgcn_wmma_f32_16x16x32_bf16(
                false, a[m].v, false, b[f].v, (short)0, acc[m][f], false, false);
}

// One wave computes a (16*MF)M x (16*NF)N tile of A[M,K] * W[N,K]^T.
// A row-major (lda), W row-major [N,K] (ldw == K stride). K % 64 == 0.
// Fragment layouts match CDNA5 ISA 7.12.2; all loads are contiguous 16B.
template<int MF, int NF>
__device__ __forceinline__ void wmma_tile(const __bf16* __restrict__ A, int lda,
                                          const __bf16* __restrict__ W, int ldw,
                                          int row0, int col0, int K,
                                          v8f (&acc)[MF][NF], int lane)
{
    const int half = lane >> 4;   // 0: lanes 0-15, 1: lanes 16-31
    const int l16  = lane & 15;

    const __bf16* ap[MF];
    const __bf16* wp[NF];
#pragma unroll
    for (int m = 0; m < MF; ++m)
        ap[m] = A + (row0 + m * 16 + l16) * lda + 8 * half;   // K {0..7,16..23}/{8..15,24..31}
#pragma unroll
    for (int f = 0; f < NF; ++f)
        wp[f] = W + (col0 + f * 16 + l16) * ldw + 8 * half;   // K {0..7,16..23}/{8..15,24..31}

    ABFrag a0[MF], a1[MF], b0[NF], b1[NF];

    load_frags(a0, ap, 0, 16);
    load_frags(b0, wp, 0, 16);

    for (int kb = 0; kb < K; kb += 64) {
        // stage chunk kb+32 while chunk kb computes
        load_frags(a1, ap, kb + 32, 16);
        load_frags(b1, wp, kb + 32, 16);
        if (kb + 96 <= K)   // gfx1250 global_prefetch_b8, two chunks ahead
            __builtin_prefetch((const void*)(ap[0] + kb + 64), 0, 1);
        do_wmma(acc, a0, b0);

        // stage chunk kb+64 while chunk kb+32 computes
        if (kb + 64 < K) {
            load_frags(a0, ap, kb + 64, 16);
            load_frags(b0, wp, kb + 64, 16);
        }
        do_wmma(acc, a1, b1);
    }
}

// ---------------------------------------------------------------------------
__global__ void k_f32_to_bf16(const float* __restrict__ in, __bf16* __restrict__ out, int n)
{
    int i = blockIdx.x * blockDim.x + threadIdx.x;
    if (i < n) out[i] = (__bf16)in[i];
}

__global__ void k_init(const float* __restrict__ h0, __bf16* __restrict__ hbuf,
                       unsigned* __restrict__ sync)
{
    int i = blockIdx.x * blockDim.x + threadIdx.x;
    if (i == 0) sync[0] = 0u;                 // reset grid-sync counter every call
    if (i < B * H) hbuf[i] = (__bf16)h0[i];   // h initial -> bf16 buffer 0
}

// xp[r, n] = sum_k x[r,k] * Wih[n,k] + b_ih[n]   (r = b*S + s, xp kept [B,S,H] f32)
__global__ void k_gemm_xp(const __bf16* __restrict__ xbf, const __bf16* __restrict__ wih,
                          const float* __restrict__ bih, float* __restrict__ xp)
{
    const int lane  = threadIdx.x & 31;
    const int gwave = (blockIdx.x * blockDim.x + threadIdx.x) >> 5;
    const int ntg   = H / 64;                 // 16 groups of 64 columns
    const int mt    = gwave / ntg;
    const int ng    = gwave % ntg;
    const int row0  = mt * 32, col0 = ng * 64;

    v8f acc[2][4] = {};
    wmma_tile<2, 4>(xbf, I, wih, I, row0, col0, I, acc, lane);

    const int half = lane >> 4, l16 = lane & 15;
#pragma unroll
    for (int mf = 0; mf < 2; ++mf)
#pragma unroll
        for (int f = 0; f < 4; ++f) {
            const int n = col0 + f * 16 + l16;
            const float bn = bih[n];
#pragma unroll
            for (int j = 0; j < 8; ++j) {
                const int m = row0 + mf * 16 + half * 8 + j;
                xp[m * H + n] = acc[mf][f][j] + bn;
            }
        }
}

// Persistent recurrence: 8 blocks x 8 waves = 64 waves, one 16x64 tile each.
// Wh (2MB bf16) stays L2-resident across all 512 steps; grid barrier via atomics.
__global__ void k_recurrence(const __bf16* __restrict__ wh, const float* __restrict__ bh,
                             const float* __restrict__ xp, __bf16* __restrict__ hbuf,
                             __bf16* __restrict__ hs, unsigned* __restrict__ sync)
{
    const int lane  = threadIdx.x & 31;
    const int gwave = (blockIdx.x * blockDim.x + threadIdx.x) >> 5; // 0..63
    const int mt    = gwave >> 4;          // 0..3   (M = 64 rows)
    const int ng    = gwave & 15;          // 0..15  (N = 1024 cols / 64)
    const int row0  = mt * 16, col0 = ng * 64;
    const unsigned nblk = gridDim.x;
    const int half = lane >> 4, l16 = lane & 15;

    for (int t = 0; t < S; ++t) {
        const __bf16* hcur  = hbuf + (t & 1) * (B * H);
        __bf16*       hnext = hbuf + ((t + 1) & 1) * (B * H);

        v8f acc[1][4] = {};
        wmma_tile<1, 4>(hcur, H, wh, H, row0, col0, H, acc, lane);

#pragma unroll
        for (int f = 0; f < 4; ++f) {
            const int n = col0 + f * 16 + l16;
            const float bn = bh[n];
#pragma unroll
            for (int j = 0; j < 8; ++j) {
                const int b = row0 + half * 8 + j;             // batch row
                float v = acc[0][f][j] + xp[(b * S + t) * H + n] + bn;
                v = fmaxf(v, 0.0f);
                const __bf16 bv = (__bf16)v;
                hnext[b * H + n]        = bv;                  // feeds step t+1
                hs[(t * B + b) * H + n] = bv;                  // [S,B,H] history
            }
        }

        // grid-wide barrier: every block arrives once per step
        __syncthreads();
        if (threadIdx.x == 0) {
            __threadfence();                                   // release h writes
            atomicAdd(sync, 1u);
            while (__hip_atomic_load(sync, __ATOMIC_RELAXED,
                                     __HIP_MEMORY_SCOPE_AGENT) < nblk * (unsigned)(t + 1)) {}
            __threadfence();                                   // acquire others' h
        }
        __syncthreads();
    }
}

// y[b,s,n] = sum_k hs[s,b,k] * V[n,k] + b_v[n]   (A rows are r = s*B + b)
__global__ void k_gemm_y(const __bf16* __restrict__ hs, const __bf16* __restrict__ vw,
                         const float* __restrict__ vb, float* __restrict__ y)
{
    const int lane  = threadIdx.x & 31;
    const int gwave = (blockIdx.x * blockDim.x + threadIdx.x) >> 5;
    const int ntg   = H / 64;
    const int mt    = gwave / ntg;
    const int ng    = gwave % ntg;
    const int row0  = mt * 32, col0 = ng * 64;

    v8f acc[2][4] = {};
    wmma_tile<2, 4>(hs, H, vw, H, row0, col0, H, acc, lane);

    const int half = lane >> 4, l16 = lane & 15;
#pragma unroll
    for (int mf = 0; mf < 2; ++mf)
#pragma unroll
        for (int f = 0; f < 4; ++f) {
            const int n = col0 + f * 16 + l16;
            const float bn = vb[n];
#pragma unroll
            for (int j = 0; j < 8; ++j) {
                const int r = row0 + mf * 16 + half * 8 + j;  // r = s*B + b
                const int b = r & (B - 1);
                const int s = r >> 6;                         // B == 64
                y[(b * S + s) * H + n] = acc[mf][f][j] + bn;
            }
        }
}

__global__ void k_hlast(const __bf16* __restrict__ hs_last, float* __restrict__ out)
{
    int i = blockIdx.x * blockDim.x + threadIdx.x;
    if (i < B * H) out[i] = (float)hs_last[i];
}

// ---------------------------------------------------------------------------
extern "C" void kernel_launch(void* const* d_in, const int* in_sizes, int n_in,
                              void* d_out, int out_size, void* d_ws, size_t ws_size,
                              hipStream_t stream)
{
    (void)in_sizes; (void)n_in; (void)out_size; (void)ws_size;

    const float* x     = (const float*)d_in[0];
    const float* h0    = (const float*)d_in[1];
    const float* wih_w = (const float*)d_in[2];
    const float* wih_b = (const float*)d_in[3];
    const float* wh_w  = (const float*)d_in[4];
    const float* wh_b  = (const float*)d_in[5];
    const float* v_w   = (const float*)d_in[6];
    const float* v_b   = (const float*)d_in[7];

    char* ws = (char*)d_ws;
    __bf16*   xbf  = (__bf16*)(ws + XBF_OFF);
    __bf16*   wih  = (__bf16*)(ws + WIH_OFF);
    __bf16*   wh   = (__bf16*)(ws + WH_OFF);
    __bf16*   vw   = (__bf16*)(ws + VW_OFF);
    float*    xp   = (float*)(ws + XP_OFF);
    __bf16*   hs   = (__bf16*)(ws + HS_OFF);
    __bf16*   hbuf = (__bf16*)(ws + HBUF_OFF);
    unsigned* sync = (unsigned*)(ws + SYNC_OFF);

    float* y = (float*)d_out;                       // [B,S,H] then h_last [B,H]
    float* hlast_out = y + (size_t)B * S * H;

    const int TB = 256;

    // 1) f32 -> bf16 conversions
    k_f32_to_bf16<<<(B * S * I + TB - 1) / TB, TB, 0, stream>>>(x, xbf, B * S * I);
    k_f32_to_bf16<<<(H * I + TB - 1) / TB, TB, 0, stream>>>(wih_w, wih, H * I);
    k_f32_to_bf16<<<(H * H + TB - 1) / TB, TB, 0, stream>>>(wh_w, wh, H * H);
    k_f32_to_bf16<<<(H * H + TB - 1) / TB, TB, 0, stream>>>(v_w, vw, H * H);

    // 2) init h buffer + sync counter
    k_init<<<(B * H + TB - 1) / TB, TB, 0, stream>>>(h0, hbuf, sync);

    // 3) input projection GEMM: waves = (B*S/32) * (H/64)
    {
        const int waves = (B * S / 32) * (H / 64);
        k_gemm_xp<<<waves * 32 / TB, TB, 0, stream>>>(xbf, wih, wih_b, xp);
    }

    // 4) persistent recurrence: 64 waves total (4 M-tiles x 16 N-groups)
    k_recurrence<<<8, TB, 0, stream>>>(wh, wh_b, xp, hbuf, hs, sync);

    // 5) output projection GEMM
    {
        const int waves = (S * B / 32) * (H / 64);
        k_gemm_y<<<waves * 32 / TB, TB, 0, stream>>>(hs, vw, v_b, y);
    }

    // 6) h_last -> f32 tail of d_out
    k_hlast<<<(B * H + TB - 1) / TB, TB, 0, stream>>>(hs + (size_t)(S - 1) * B * H, hlast_out);
}
